// GNN_62895501083190
// MI455X (gfx1250) — matrix operand order repaired and checked by reading, not compiled, not run on hardware
//
#include <hip/hip_runtime.h>

// ---------------- problem constants ----------------
#define N_NODES 100000
#define N_EDGES 1600000
#define F_IN    512
#define HID     156
#define N_CLS   7
#define NT      10            // 10 column tiles of 16 (160 >= 156, tail masked)
#define KT      (F_IN / 32)   // 16 k-steps of 32 (bf16 WMMA K), power of two
#define BFRAG_ELEMS (KT * NT * 32 * 16)        // 81920 bf16
#define BFRAG_BYTES (BFRAG_ELEMS * 2)          // 163840 B = 160 KB (fits 320 KB WGP LDS)

typedef __attribute__((ext_vector_type(16))) __bf16 v16bf;
typedef __attribute__((ext_vector_type(8)))  float  v8f;

// ---------------- degree / symmetric norm ----------------
__global__ void deg_init_kernel(float* dis) {
    int i = blockIdx.x * blockDim.x + threadIdx.x;
    if (i < N_NODES) dis[i] = 1.0f;                 // self-loop
}
__global__ void deg_edges_kernel(const int* __restrict__ dst, float* dis) {
    int e = blockIdx.x * blockDim.x + threadIdx.x;
    if (e < N_EDGES) unsafeAtomicAdd(&dis[dst[e]], 1.0f);
}
__global__ void deg_rsqrt_kernel(float* dis) {
    int i = blockIdx.x * blockDim.x + threadIdx.x;
    if (i < N_NODES) dis[i] = rsqrtf(dis[i]);
}

// ---------------- repack W1 -> bf16 B-fragments ----------------
// Layout per 32x16 bf16 B tile (K x N): lane L holds col N = L&15,
// K run = (L>>4)*16 + j (j=0..15, even/odd pairs per VGPR).
// Bbuf[((kk*NT + t)*32 + lane)*16 + j]  => per-lane contiguous 32 bytes.
__global__ void bprep_kernel(const float* __restrict__ W1, __bf16* __restrict__ Bbuf) {
    int idx = blockIdx.x * blockDim.x + threadIdx.x;   // idx = kk*320 + t*32 + lane
    if (idx >= KT * NT * 32) return;
    int lane = idx & 31;
    int t    = (idx >> 5) % NT;
    int kk   = idx / (NT * 32);
    int n    = t * 16 + (lane & 15);
    int kb   = kk * 32 + (lane >> 4) * 16;
    __bf16* out = Bbuf + (size_t)idx * 16;
#pragma unroll
    for (int j = 0; j < 16; ++j) {
        float v = (n < HID) ? W1[(size_t)(kb + j) * HID + n] : 0.0f;
        out[j] = (__bf16)v;
    }
}

// ---------------- GEMM1: H = X @ W1 via bf16 WMMA, fp32 accum ----------------
// One wave -> 16 rows x 160 cols. 8 waves per block -> 128 rows/block.
// Whole repacked W1 (160 KB) staged in LDS once per block. A tile is double-
// buffered in registers with an unconditional wrapped-index prefetch (KT is a
// power of two) so the loop body stays uniform (no peeling, no spills) and the
// global load latency overlaps the 10-WMMA chain. B fragments are loaded
// just-in-time from LDS; the scheduler pipelines the ds_load_b128 pairs.
__global__ void gemm1_wmma_kernel(const float* __restrict__ X,
                                  const __bf16* __restrict__ Bbuf,
                                  float* __restrict__ H) {
    extern __shared__ __bf16 Bs[];                 // BFRAG_BYTES dynamic LDS

    // cooperative stage: 163840 B = 10240 float4, 40 per thread
    {
        const float4* s4 = (const float4*)Bbuf;
        float4* d4 = (float4*)Bs;
        for (int i = threadIdx.x; i < BFRAG_BYTES / 16; i += blockDim.x)
            d4[i] = s4[i];
    }
    __syncthreads();

    const int lane = threadIdx.x & 31;
    const int wave = threadIdx.x >> 5;
    const int half = lane >> 4;
    const int l15  = lane & 15;
    const int rowBase = (blockIdx.x * 8 + wave) * 16;
    if (rowBase >= N_NODES) return;

    v8f acc[NT];
#pragma unroll
    for (int t = 0; t < NT; ++t) acc[t] = (v8f){0.f,0.f,0.f,0.f,0.f,0.f,0.f,0.f};

    const float* xrow = X + (size_t)(rowBase + l15) * F_IN;   // A row M = lane&15

    // prologue: load A tile for kk = 0
    // A 16x32 bf16 fragment: lane holds K run [k0..k0+15], k0 = half*16
    float4 a0 = *(const float4*)(xrow + half * 16);
    float4 a1 = *(const float4*)(xrow + half * 16 + 4);
    float4 a2 = *(const float4*)(xrow + half * 16 + 8);
    float4 a3 = *(const float4*)(xrow + half * 16 + 12);

    for (int kk = 0; kk < KT; ++kk) {
        float4 c0 = a0, c1 = a1, c2 = a2, c3 = a3;
        // unconditional prefetch of next A tile (wraps to tile 0 on last iter)
        const int kn = ((kk + 1) & (KT - 1)) * 32 + half * 16;
        a0 = *(const float4*)(xrow + kn);
        a1 = *(const float4*)(xrow + kn + 4);
        a2 = *(const float4*)(xrow + kn + 8);
        a3 = *(const float4*)(xrow + kn + 12);

        v16bf a;
        a[0]=(__bf16)c0.x;  a[1]=(__bf16)c0.y;  a[2]=(__bf16)c0.z;  a[3]=(__bf16)c0.w;
        a[4]=(__bf16)c1.x;  a[5]=(__bf16)c1.y;  a[6]=(__bf16)c1.z;  a[7]=(__bf16)c1.w;
        a[8]=(__bf16)c2.x;  a[9]=(__bf16)c2.y;  a[10]=(__bf16)c2.z; a[11]=(__bf16)c2.w;
        a[12]=(__bf16)c3.x; a[13]=(__bf16)c3.y; a[14]=(__bf16)c3.z; a[15]=(__bf16)c3.w;

        // just-in-time B fragments from LDS + WMMA chain (scheduler pipelines)
#pragma unroll
        for (int t = 0; t < NT; ++t) {
            v16bf b = *(const v16bf*)(Bs + (((size_t)kk * NT + t) * 32 + lane) * 16);
            acc[t] = __builtin_amdgcn_wmma_f32_16x16x32_bf16(
                false, a, false, b, (short)0, acc[t], false, false);
        }
    }
    // C/D layout: VGPR r, lanes 0-15 -> M=r, N=lane; lanes 16-31 -> M=r+8, N=lane-16
#pragma unroll
    for (int t = 0; t < NT; ++t) {
        const int col = t * 16 + l15;
        if (col < HID) {
#pragma unroll
            for (int r = 0; r < 8; ++r) {
                const int row = rowBase + r + half * 8;
                H[(size_t)row * HID + col] = acc[t][r];
            }
        }
    }
}

// ---------------- aggregation layer 1 ----------------
__global__ void agg1_init_kernel(const float* __restrict__ H,
                                 const float* __restrict__ dis,
                                 float* __restrict__ agg) {
    size_t idx = (size_t)blockIdx.x * blockDim.x + threadIdx.x;
    if (idx >= (size_t)N_NODES * HID) return;
    int row = (int)(idx / HID);
    float d = dis[row];
    agg[idx] = H[idx] * d * d;                      // self-loop: 1/deg
}
// one wave per edge, lanes sweep 156 channels
__global__ void agg1_edges_kernel(const int* __restrict__ src,
                                  const int* __restrict__ dst,
                                  const float* __restrict__ dis,
                                  const float* __restrict__ H,
                                  float* __restrict__ agg) {
    int e    = blockIdx.x * 8 + (threadIdx.x >> 5);
    int lane = threadIdx.x & 31;
    if (e >= N_EDGES) return;
    int s = src[e], d = dst[e];
    float w = dis[s] * dis[d];
    const float* hs = H + (size_t)s * HID;
    float* ad = agg + (size_t)d * HID;
    for (int c = lane; c < HID; c += 32)
        unsafeAtomicAdd(&ad[c], w * hs[c]);
}

// ---------------- GEMM2 fused bias+ReLU: Z = relu(agg+b1) @ W2 ----------------
__global__ void gemm2_kernel(const float* __restrict__ agg,
                             const float* __restrict__ b1,
                             const float* __restrict__ W2,
                             float* __restrict__ Z) {
    int n    = blockIdx.x * 8 + (threadIdx.x >> 5);
    int lane = threadIdx.x & 31;
    if (n >= N_NODES) return;
    float acc[N_CLS];
#pragma unroll
    for (int j = 0; j < N_CLS; ++j) acc[j] = 0.0f;
    const float* row = agg + (size_t)n * HID;
    for (int k = lane; k < HID; k += 32) {
        float hv = fmaxf(row[k] + b1[k], 0.0f);
#pragma unroll
        for (int j = 0; j < N_CLS; ++j) acc[j] += hv * W2[k * N_CLS + j];
    }
#pragma unroll
    for (int j = 0; j < N_CLS; ++j)
        for (int off = 16; off > 0; off >>= 1)
            acc[j] += __shfl_down(acc[j], off, 32);
    if (lane == 0) {
#pragma unroll
        for (int j = 0; j < N_CLS; ++j) Z[(size_t)n * N_CLS + j] = acc[j];
    }
}

// ---------------- aggregation layer 2 (into d_out) ----------------
__global__ void agg2_init_kernel(const float* __restrict__ Z,
                                 const float* __restrict__ dis,
                                 float* __restrict__ out) {
    int idx = blockIdx.x * blockDim.x + threadIdx.x;
    if (idx >= N_NODES * N_CLS) return;
    float d = dis[idx / N_CLS];
    out[idx] = Z[idx] * d * d;
}
__global__ void agg2_edges_kernel(const int* __restrict__ src,
                                  const int* __restrict__ dst,
                                  const float* __restrict__ dis,
                                  const float* __restrict__ Z,
                                  float* __restrict__ out) {
    int e = blockIdx.x * blockDim.x + threadIdx.x;
    if (e >= N_EDGES) return;
    int s = src[e], d = dst[e];
    float w = dis[s] * dis[d];
    const float* zs = Z + (size_t)s * N_CLS;
    float* od = out + (size_t)d * N_CLS;
#pragma unroll
    for (int j = 0; j < N_CLS; ++j)
        unsafeAtomicAdd(&od[j], w * zs[j]);
}

// ---------------- bias + log_softmax (in place on d_out) ----------------
__global__ void logsoftmax_kernel(const float* __restrict__ b2, float* __restrict__ out) {
    int n = blockIdx.x * blockDim.x + threadIdx.x;
    if (n >= N_NODES) return;
    float v[N_CLS];
    float m = -1e30f;
#pragma unroll
    for (int j = 0; j < N_CLS; ++j) {
        v[j] = out[(size_t)n * N_CLS + j] + b2[j];
        m = fmaxf(m, v[j]);
    }
    float s = 0.0f;
#pragma unroll
    for (int j = 0; j < N_CLS; ++j) s += expf(v[j] - m);
    float l = logf(s);
#pragma unroll
    for (int j = 0; j < N_CLS; ++j) out[(size_t)n * N_CLS + j] = v[j] - m - l;
}

// ---------------- launcher ----------------
extern "C" void kernel_launch(void* const* d_in, const int* in_sizes, int n_in,
                              void* d_out, int out_size, void* d_ws, size_t ws_size,
                              hipStream_t stream) {
    const float* X    = (const float*)d_in[0];
    const int*   eidx = (const int*)d_in[1];
    const int*   src  = eidx;
    const int*   dst  = eidx + N_EDGES;
    const float* W1   = (const float*)d_in[2];
    const float* b1   = (const float*)d_in[3];
    const float* W2   = (const float*)d_in[4];
    const float* b2   = (const float*)d_in[5];
    float*       out  = (float*)d_out;

    // workspace carve-up (256B aligned)
    char* ws = (char*)d_ws;
    size_t off = 0;
    auto carve = [&](size_t bytes) { void* p = ws + off; off = (off + bytes + 255) & ~(size_t)255; return p; };
    float*  dis  = (float*)carve((size_t)N_NODES * 4);
    __bf16* Bbuf = (__bf16*)carve((size_t)BFRAG_BYTES);
    float*  H    = (float*)carve((size_t)N_NODES * HID * 4);
    float*  agg  = (float*)carve((size_t)N_NODES * HID * 4);
    float*  Z    = (float*)carve((size_t)N_NODES * N_CLS * 4);
    (void)ws_size; (void)n_in; (void)in_sizes; (void)out_size;

    const int T = 256;
    // norm
    deg_init_kernel <<<(N_NODES + T - 1) / T, T, 0, stream>>>(dis);
    deg_edges_kernel<<<(N_EDGES + T - 1) / T, T, 0, stream>>>(dst, dis);
    deg_rsqrt_kernel<<<(N_NODES + T - 1) / T, T, 0, stream>>>(dis);
    // W1 repack -> B fragments
    bprep_kernel<<<(KT * NT * 32 + T - 1) / T, T, 0, stream>>>(W1, Bbuf);
    // layer 1 GEMM (WMMA): 8 waves/block, 16 rows/wave, 160 KB LDS-staged B
    gemm1_wmma_kernel<<<(N_NODES / 16 + 7) / 8, T, BFRAG_BYTES, stream>>>(X, Bbuf, H);
    // layer 1 aggregation
    agg1_init_kernel <<<(int)(((size_t)N_NODES * HID + T - 1) / T), T, 0, stream>>>(H, dis, agg);
    agg1_edges_kernel<<<(N_EDGES + 7) / 8, T, 0, stream>>>(src, dst, dis, H, agg);
    // layer 2 GEMM fused bias+relu
    gemm2_kernel<<<(N_NODES + 7) / 8, T, 0, stream>>>(agg, b1, W2, Z);
    // layer 2 aggregation into d_out
    agg2_init_kernel <<<(N_NODES * N_CLS + T - 1) / T, T, 0, stream>>>(Z, dis, out);
    agg2_edges_kernel<<<(N_EDGES + T - 1) / T, T, 0, stream>>>(src, dst, dis, Z, out);
    // bias + log_softmax in place
    logsoftmax_kernel<<<(N_NODES + T - 1) / T, T, 0, stream>>>(b2, out);
}
